// FromStatesToVoice_20134806684406
// MI455X (gfx1250) — matrix operand order repaired and checked by reading, not compiled, not run on hardware
//
#include <hip/hip_runtime.h>
#include <hip/hip_bf16.h>

// ---------------------------------------------------------------------------
// Problem constants (from reference): L=3, SH=256, H=50, B=2048, T=256.
// Padded dims: HP=64 (K and per-gate N padded) -> 4 WMMA N-tiles per gate.
// ---------------------------------------------------------------------------

typedef _Float16 v16h __attribute__((ext_vector_type(16)));
typedef float    v8f  __attribute__((ext_vector_type(8)));
typedef float    f32x4 __attribute__((ext_vector_type(4)));

#define Lr   3
#define SHr  256
#define Hr   50
#define Br   2048
#define HP   64          // padded hidden (K and per-gate N)
#define WMAT 4096        // halves per padded weight matrix (64x64)

#if __has_builtin(__builtin_amdgcn_sched_barrier)
#define SCHED_FENCE() __builtin_amdgcn_sched_barrier(0x407) // ALU may cross, mem may not
#else
#define SCHED_FENCE()
#endif

// ---------------------------------------------------------------------------
// WMMA helper: D = A(16x32 f16) x B(32x16 f16) + C(f32)
// ---------------------------------------------------------------------------
static __device__ __forceinline__ v8f wmma_f16(v16h a, v16h b, v8f c) {
  return __builtin_amdgcn_wmma_f32_16x16x32_f16(false, a, false, b,
                                                (short)0, c, false, false);
}

// fast sigmoid / tanh via v_exp_f32 + v_rcp_f32
static __device__ __forceinline__ float fsig(float x) {
  float e = __builtin_amdgcn_exp2f(-1.44269504f * x);
  return __builtin_amdgcn_rcpf(1.0f + e);
}
static __device__ __forceinline__ float ftanh(float x) {
  return 2.0f * fsig(2.0f * x) - 1.0f;
}

// A operand (16x32 f16): p already includes per-lane offset
// (row = lane&15)*HP + (lane>>4)*8; chunk c adds c*32; second half at +16.
static __device__ __forceinline__ v16h load_a16(const _Float16* p, int coff) {
  union { v16h v; f32x4 f[2]; } u;
  u.f[0] = *(const f32x4*)(p + coff);
  u.f[1] = *(const f32x4*)(p + coff + 16);
  return u.v;
}

// B operand (32x16 f16): p already includes per-lane offset
// (n = lane&15)*HP + (lane>>4)*8; second half at +16; caller adds matrix/tile/chunk const offset.
static __device__ __forceinline__ v16h load_b16(const _Float16* p, int off) {
  union { v16h v; f32x4 f[2]; } u;
  u.f[0] = *(const f32x4*)(p + off);
  u.f[1] = *(const f32x4*)(p + off + 16);
  return u.v;
}

// ---------------------------------------------------------------------------
// Prep kernel 1: convert GRU weights to padded f16 B-operand layout.
// wbuf[(l*2+mat)*3+g][n:64][k:64], n/k >= 50 zero padded.
// ---------------------------------------------------------------------------
__global__ __launch_bounds__(256) void prep_weights(const float* __restrict__ wih,
                                                    const float* __restrict__ whh,
                                                    _Float16* __restrict__ wbuf) {
  const int m   = blockIdx.x;          // 0..17
  const int l   = m / 6;
  const int mat = (m / 3) % 2;
  const int g   = m % 3;
  const float* src = (mat ? whh : wih) + l * (3 * Hr * Hr) + g * Hr * Hr;
  _Float16* dst = wbuf + m * WMAT;
  for (int i = threadIdx.x; i < WMAT; i += 256) {
    const int n = i >> 6, k = i & 63;
    float v = (n < Hr && k < Hr) ? src[n * Hr + k] : 0.0f;
    dst[i] = (_Float16)v;
  }
}

// ---------------------------------------------------------------------------
// Prep kernel 2: states0[l][b][0:64] = tanh(next.cw+cb) - tanh(prev.cw+cb)
// ---------------------------------------------------------------------------
__global__ __launch_bounds__(64) void prep_states(const float* __restrict__ prev,
                                                  const float* __restrict__ next,
                                                  const float* __restrict__ cw,
                                                  const float* __restrict__ cb,
                                                  float* __restrict__ sbuf) {
  __shared__ float rp[SHr], rn[SHr];
  const int bid = blockIdx.x;                 // 0 .. 3*2048-1
  const float* pp = prev + (size_t)bid * SHr;
  const float* pn = next + (size_t)bid * SHr;
  for (int i = threadIdx.x; i < SHr; i += 64) { rp[i] = pp[i]; rn[i] = pn[i]; }
  __syncthreads();
  const int j = threadIdx.x;                  // 0..63
  float out = 0.0f;
  if (j < Hr) {
    const float* c = cw + j * SHr;
    float ap = cb[j], an = cb[j];
#pragma unroll 8
    for (int i = 0; i < SHr; ++i) { float cv = c[i]; ap += cv * rp[i]; an += cv * rn[i]; }
    out = tanhf(an) - tanhf(ap);
  }
  sbuf[(size_t)bid * HP + j] = out;
}

// ---------------------------------------------------------------------------
// Main persistent GRU kernel. 32 blocks x 128 threads (4 independent waves).
// Wave w owns rows b0 = blockIdx*64 + w*16 for all T steps.
// f16 hidden state lives in per-wave LDS (master copy = WMMA input copy).
// ---------------------------------------------------------------------------
__global__ __launch_bounds__(128, 1) void rnn_main(
    const _Float16* __restrict__ wbuf,   // [18][64][64] f16 padded weights
    const float* __restrict__ sbuf,      // [3][2048][64] f32 initial states
    const float* __restrict__ bih,       // [3][150]
    const float* __restrict__ bhh,       // [3][150]
    const float* __restrict__ s0w,       // [50]
    const float* __restrict__ s0b,       // [50]
    const float* __restrict__ s1w,       // [50]
    const float* __restrict__ s1b,       // [1]
    float* __restrict__ out,             // [T][2048]
    int T) {
  __shared__ _Float16 xbufS[4][16 * HP];          // per-wave layer-0 input
  __shared__ _Float16 hbufS[4][Lr][16 * HP];      // per-wave f16 hidden state
  __shared__ float    vscrS[4][16];               // per-wave voice broadcast
  __shared__ float    biasI[Lr][3][HP];           // b_ih padded
  __shared__ float    biasH[Lr][3][HP];           // b_hh padded
  __shared__ float    s0wS[HP], s0bS[HP], s1wS[HP];
  __shared__ float    s1bS;

  const int tid  = threadIdx.x;
  const int lane = tid & 31;
  const int w    = tid >> 5;
  const int hi   = lane >> 4;
  const int lo   = lane & 15;
  const int b0   = blockIdx.x * 64 + w * 16;

  // --- block-cooperative init of biases / scales into LDS -------------------
  for (int i = tid; i < Lr * 3 * HP; i += 128) {
    const int l = i / (3 * HP), r = i % (3 * HP), g = r / HP, col = r % HP;
    float vI = 0.0f, vH = 0.0f;
    if (col < Hr) {
      vI = bih[l * 3 * Hr + g * Hr + col];
      vH = bhh[l * 3 * Hr + g * Hr + col];
    }
    biasI[l][g][col] = vI;
    biasH[l][g][col] = vH;
  }
  if (tid < HP) {
    s0wS[tid] = (tid < Hr) ? s0w[tid] : 0.0f;
    s0bS[tid] = (tid < Hr) ? s0b[tid] : 0.0f;
    s1wS[tid] = (tid < Hr) ? s1w[tid] : 0.0f;
  }
  if (tid == 0) s1bS = s1b[0];

  // prefetch the staged weight region (hot for all T steps)
  {
    const char* wbp = (const char*)wbuf;
    for (int off = tid * 128; off < 18 * WMAT * 2; off += 128 * 128)
      __builtin_prefetch(wbp + off, 0, 0);
  }
  __syncthreads();

  // --- per-wave pointers / per-lane base offsets ----------------------------
  _Float16* xw  = xbufS[w];
  _Float16* hw0 = hbufS[w][0];
  _Float16* hw1 = hbufS[w][1];
  _Float16* hw2 = hbufS[w][2];
  _Float16* hws[Lr] = { hw0, hw1, hw2 };

  const int aoff = lo * HP + hi * 8;        // A operand per-lane base (halves)
  const int hoff = hi * 8 * HP + lo;        // C/D-layout per-lane base (halves)

  // --- initial hidden state: f32 states0 -> f16 LDS -------------------------
#pragma unroll
  for (int l = 0; l < Lr; ++l) {
    _Float16* hS = hws[l] + hoff;
#pragma unroll
    for (int tl = 0; tl < 4; ++tl) {
#pragma unroll
      for (int v = 0; v < 8; ++v) {
        const int row = b0 + v + hi * 8;
        hS[v * HP + tl * 16] =
            (_Float16)sbuf[((size_t)l * Br + row) * HP + lo + tl * 16];
      }
    }
  }
  // x for t=0: voice = 0  =>  x[m][j] = s0b[j]
  {
    const int cb = hi * 32;
#pragma unroll 8
    for (int j = 0; j < 32; ++j)
      xw[lo * HP + cb + j] = (_Float16)(s0bS[cb + j]);
  }

  const v8f zf = {};   // zero C operand (lowers to inline 0)

  // --- time loop ------------------------------------------------------------
  for (int t = 0; t < T; ++t) {
    // Opaque *offset* (not pointer): blocks LICM of the 288 weight loads out
    // of the time loop while keeping global address-space provenance, so the
    // compiler emits global_load_b128 (LOADcnt only) instead of flat_load
    // (which would couple LOADcnt with the LDS DScnt pipeline).
    int wofs = 0;
    asm volatile("" : "+v"(wofs));
    const _Float16* wB = wbuf + (lo * HP + hi * 8 + wofs);

    const _Float16* xsrc = xw;
#pragma unroll
    for (int l = 0; l < Lr; ++l) {
      const v16h axA0 = load_a16(xsrc + aoff, 0);
      const v16h axA1 = load_a16(xsrc + aoff, 32);
      const v16h ahA0 = load_a16(hws[l] + aoff, 0);
      const v16h ahA1 = load_a16(hws[l] + aoff, 32);

      const int mIH = (l * 2 + 0) * 3;   // matrices: r,z,n (input)
      const int mHH = (l * 2 + 1) * 3;   // matrices: r,z,n (recurrent)

#pragma unroll
      for (int tl = 0; tl < 4; ++tl) {
        const int to = tl * 16 * HP;     // tile offset inside a matrix

        v8f ar = wmma_f16(axA0, load_b16(wB, (mIH + 0) * WMAT + to + 0),  zf);
        ar     = wmma_f16(axA1, load_b16(wB, (mIH + 0) * WMAT + to + 32), ar);
        ar     = wmma_f16(ahA0, load_b16(wB, (mHH + 0) * WMAT + to + 0),  ar);
        ar     = wmma_f16(ahA1, load_b16(wB, (mHH + 0) * WMAT + to + 32), ar);

        v8f az = wmma_f16(axA0, load_b16(wB, (mIH + 1) * WMAT + to + 0),  zf);
        az     = wmma_f16(axA1, load_b16(wB, (mIH + 1) * WMAT + to + 32), az);
        az     = wmma_f16(ahA0, load_b16(wB, (mHH + 1) * WMAT + to + 0),  az);
        az     = wmma_f16(ahA1, load_b16(wB, (mHH + 1) * WMAT + to + 32), az);

        v8f ai = wmma_f16(axA0, load_b16(wB, (mIH + 2) * WMAT + to + 0),  zf);
        ai     = wmma_f16(axA1, load_b16(wB, (mIH + 2) * WMAT + to + 32), ai);

        v8f ag = wmma_f16(ahA0, load_b16(wB, (mHH + 2) * WMAT + to + 0),  zf);
        ag     = wmma_f16(ahA1, load_b16(wB, (mHH + 2) * WMAT + to + 32), ag);

        // gate math; biases folded here (accumulators started at 0)
        const int col = lo + tl * 16;
        const float br = biasI[l][0][col] + biasH[l][0][col];
        const float bz = biasI[l][1][col] + biasH[l][1][col];
        const float bi = biasI[l][2][col];
        const float bh = biasH[l][2][col];

        _Float16* hS = hws[l] + hoff;
#pragma unroll
        for (int v = 0; v < 8; ++v) {
          const float hold = (float)hS[v * HP + tl * 16];
          const float r = fsig(ar[v] + br);
          const float z = fsig(az[v] + bz);
          const float n = ftanh(ai[v] + bi + r * (ag[v] + bh));
          hS[v * HP + tl * 16] = (_Float16)(n + z * (hold - n));
        }
        SCHED_FENCE();
      }
      xsrc = hws[l];   // next layer's input
    }

    // voice_out = h[2] @ s1w + s1b  (lanes 0..15 each own one batch row)
    if (lane < 16) {
      const _Float16* hr = hw2 + lane * HP;
      float acc = s1bS;
#pragma unroll 8
      for (int j = 0; j < HP; ++j) acc += (float)hr[j] * s1wS[j];
      out[(size_t)t * Br + b0 + lane] = acc;
      vscrS[w][lane] = acc;
    }
    // x[t+1] = voice * s0w + s0b  (all 32 lanes; DS in-order within wave)
    {
      const int cb = hi * 32;
      const float vv = vscrS[w][lo];
#pragma unroll 8
      for (int j = 0; j < 32; ++j)
        xw[lo * HP + cb + j] = (_Float16)(vv * s0wS[cb + j] + s0bS[cb + j]);
    }
  }
}

// ---------------------------------------------------------------------------
// Launcher
// ---------------------------------------------------------------------------
extern "C" void kernel_launch(void* const* d_in, const int* in_sizes, int n_in,
                              void* d_out, int out_size, void* d_ws, size_t ws_size,
                              hipStream_t stream) {
  const float* prev = (const float*)d_in[0];
  const float* next = (const float*)d_in[1];
  const float* cw   = (const float*)d_in[2];
  const float* cb   = (const float*)d_in[3];
  const float* s0w  = (const float*)d_in[4];
  const float* s0b  = (const float*)d_in[5];
  const float* s1w  = (const float*)d_in[6];
  const float* s1b  = (const float*)d_in[7];
  const float* wih  = (const float*)d_in[8];
  const float* whh  = (const float*)d_in[9];
  const float* bih  = (const float*)d_in[10];
  const float* bhh  = (const float*)d_in[11];

  _Float16* wbuf = (_Float16*)d_ws;                           // 147456 bytes
  float*    sbuf = (float*)((char*)d_ws + 18 * WMAT * 2);     // 1572864 bytes

  const int T = out_size / Br;   // 256

  prep_weights<<<18, 256, 0, stream>>>(wih, whh, wbuf);
  prep_states<<<Lr * Br, 64, 0, stream>>>(prev, next, cw, cb, sbuf);
  rnn_main<<<32, 128, 0, stream>>>(wbuf, sbuf, bih, bhh,
                                   s0w, s0b, s1w, s1b,
                                   (float*)d_out, T);
  (void)in_sizes; (void)n_in; (void)ws_size;
}
